// SubgraphGNNLSTM_45397804318727
// MI455X (gfx1250) — compile-verified, matrix operand order checked by reading, not run on hardware
//
#include <hip/hip_runtime.h>
#include <hip/hip_bf16.h>

typedef __attribute__((ext_vector_type(16))) __bf16 v16bf;
typedef __attribute__((ext_vector_type(8)))  __bf16 v8bf;
typedef __attribute__((ext_vector_type(8)))  float  v8f;

#define NN 50000
#define EE 300000
#define ETOT 350000   // E + N self loops
#define HH 256
#define NEGV (-1e9f)

// ---------------- helpers ----------------
__device__ __forceinline__ float wave_sum(float v) {
#pragma unroll
  for (int off = 16; off > 0; off >>= 1) v += __shfl_xor(v, off, 32);
  return v;
}

__device__ __forceinline__ void atomicMaxF(float* addr, float val) {
  if (val >= 0.f)
    atomicMax((int*)addr, __float_as_int(val));
  else
    atomicMin((unsigned int*)addr, __float_as_uint(val));
}

__device__ __forceinline__ v16bf cat16(v8bf lo, v8bf hi) {
  return __builtin_shufflevector(lo, hi, 0,1,2,3,4,5,6,7,8,9,10,11,12,13,14,15);
}

// ---------------- conversion kernels ----------------
__global__ void k_cvt_bf16(const float* __restrict__ in, __bf16* __restrict__ out, int n) {
  int i = blockIdx.x * 256 + threadIdx.x;
  if (i < n) out[i] = (__bf16)in[i];
}

// Wt[n*256+k] = bf16(W[k*256+n])  (N-major so B fragments are contiguous in K)
__global__ void k_prep_wt(const float* __restrict__ W, __bf16* __restrict__ Wt) {
  int k = blockIdx.x;       // 0..255
  int n = threadIdx.x;      // 0..255
  Wt[n * 256 + k] = (__bf16)W[k * 256 + n];
}

__global__ void k_init_m(float* __restrict__ m, int n) {
  int i = blockIdx.x * 256 + threadIdx.x;
  if (i < n) m[i] = NEGV;
}

__global__ void k_fix_m(float* __restrict__ m, int n) {
  int i = blockIdx.x * 256 + threadIdx.x;
  if (i < n) { float v = m[i]; m[i] = (v <= NEGV * 0.5f) ? 0.f : v; }
}

// ---------------- WMMA GEMM: C[M,256] = A_bf16[M,256] x Wt_bf16(N-major 256x256) -------------
// Block: 256 threads = 8 waves. Tile: 64 rows x 128 cols. blockIdx.y picks the column half.
// LDS: A tile 64x256 bf16 + W tile 128x256 bf16, both padded -> ~99 KB, all inner-loop
// operands come from LDS (ds_load), no global latency inside the WMMA chain.
#define GM_ROWS 64
#define GN_COLS 128
#define LDA 264   // bf16 elements per row in LDS (16B padding -> conflict-free)

__global__ __launch_bounds__(256) void k_gemm_bf16(
    const __bf16* __restrict__ A, const __bf16* __restrict__ Wt,
    float* __restrict__ C, int M) {
  __shared__ __bf16 As[GM_ROWS * LDA];   // 33,792 B
  __shared__ __bf16 Ws[GN_COLS * LDA];   // 67,584 B
  const int t = threadIdx.x;
  const int block_row = blockIdx.x * GM_ROWS;
  const int nbase = blockIdx.y * GN_COLS;

  // stage A tile (64 x 256 bf16): 2048 chunks of 16B
#pragma unroll
  for (int c = 0; c < 8; c++) {
    int chunk = t + 256 * c;
    int row = chunk >> 5;
    int koff = (chunk & 31) * 8;
    int gr = block_row + row;
    if (gr >= M) gr = M - 1;             // clamp (stores are guarded)
    uint4 v = *(const uint4*)(A + (size_t)gr * HH + koff);
    *(uint4*)(&As[row * LDA + koff]) = v;
  }
  // stage W tile (128 x 256 bf16): 4096 chunks of 16B
#pragma unroll
  for (int c = 0; c < 16; c++) {
    int chunk = t + 256 * c;
    int row = chunk >> 5;
    int koff = (chunk & 31) * 8;
    uint4 v = *(const uint4*)(Wt + (size_t)(nbase + row) * HH + koff);
    *(uint4*)(&Ws[row * LDA + koff]) = v;
  }
  __syncthreads();

  const int wave = t >> 5;
  const int lane = t & 31;
  const int half = lane >> 4;
  const int idx  = lane & 15;
  const int rbase  = (wave & 3) * 16;    // row group within tile
  const int ntbase = (wave >> 2) * 4;    // N-tile group (8 tiles of 16 cols split 2 ways)
  const int mrow = rbase + idx;

  // preload all 8 K-step A fragments from LDS
  v16bf afr[8];
#pragma unroll
  for (int kt = 0; kt < 8; kt++) {
    int k0 = kt * 32 + half * 8;
    v8bf lo = *(const v8bf*)(&As[mrow * LDA + k0]);
    v8bf hi = *(const v8bf*)(&As[mrow * LDA + k0 + 16]);
    afr[kt] = cat16(lo, hi);
  }

  const bool full = (block_row + GM_ROWS) <= M;

#pragma unroll
  for (int nt = 0; nt < 4; nt++) {
    const int ntg = ntbase + nt;
    const int wrow = ntg * 16 + idx;     // row of Ws (output column within 128-col tile)
    // cluster the 8 B-fragment LDS loads, then run the WMMA chain
    v16bf bfr[8];
#pragma unroll
    for (int kt = 0; kt < 8; kt++) {
      int k0 = kt * 32 + half * 8;
      v8bf lo = *(const v8bf*)(&Ws[wrow * LDA + k0]);
      v8bf hi = *(const v8bf*)(&Ws[wrow * LDA + k0 + 16]);
      bfr[kt] = cat16(lo, hi);
    }
    v8f acc = {0.f, 0.f, 0.f, 0.f, 0.f, 0.f, 0.f, 0.f};
#pragma unroll
    for (int kt = 0; kt < 8; kt++) {
      acc = __builtin_amdgcn_wmma_f32_16x16x32_bf16(
          false, afr[kt], false, bfr[kt], (short)0, acc, false, false);
    }
    const int gcol = nbase + ntg * 16 + idx;
    if (full) {
#pragma unroll
      for (int r = 0; r < 8; r++) {
        int grow = block_row + rbase + r + 8 * half;
        C[(size_t)grow * HH + gcol] = acc[r];
      }
    } else {
#pragma unroll
      for (int r = 0; r < 8; r++) {
        int grow = block_row + rbase + r + 8 * half;
        if (grow < M) C[(size_t)grow * HH + gcol] = acc[r];
      }
    }
  }
}

// ---------------- s,d = hw . a_src / a_dst (wave per node) ----------------
__global__ __launch_bounds__(256) void k_sd(
    const float* __restrict__ hw, const float* __restrict__ asrc,
    const float* __restrict__ adst, float* __restrict__ S, float* __restrict__ D, int n) {
  int wave = (blockIdx.x * 256 + threadIdx.x) >> 5;
  int lane = threadIdx.x & 31;
  if (wave >= n) return;
  const float* row = hw + (size_t)wave * HH;
  float s = 0.f, d = 0.f;
#pragma unroll
  for (int j = 0; j < 8; j++) {
    int f = lane + 32 * j;
    float v = row[f];
    s += v * asrc[f];
    d += v * adst[f];
  }
  s = wave_sum(s); d = wave_sum(d);
  if (lane == 0) { S[wave] = s; D[wave] = d; }
}

// ---------------- edge pass 1: segment max (thread per edge) ----------------
__global__ void k_edge_max(const int* __restrict__ ei, const unsigned char* __restrict__ mask,
                           const float* __restrict__ S, const float* __restrict__ D,
                           float* __restrict__ m) {
  int idx = blockIdx.x * 256 + threadIdx.x;
  if (idx >= ETOT) return;
  int s_, d_; bool val;
  if (idx < EE) { s_ = ei[idx]; d_ = ei[EE + idx]; val = mask[s_] && mask[d_]; }
  else          { s_ = d_ = idx - EE; val = mask[s_] != 0; }
  if (!val) return;
  float e = S[s_] + D[d_];
  e = (e >= 0.f) ? e : 0.2f * e;
  atomicMaxF(&m[d_], e);
}

// ---------------- edge pass 2: exp, den, acc += ex * hw[src] (wave per edge) ----------------
__global__ __launch_bounds__(256) void k_edge_acc(
    const int* __restrict__ ei, const unsigned char* __restrict__ mask,
    const float* __restrict__ S, const float* __restrict__ D, const float* __restrict__ m,
    const float* __restrict__ hw, float* __restrict__ den, float* __restrict__ acc) {
  int wave = (blockIdx.x * 256 + threadIdx.x) >> 5;
  int lane = threadIdx.x & 31;
  if (wave >= ETOT) return;
  int s_, d_; bool val;
  if (wave < EE) { s_ = ei[wave]; d_ = ei[EE + wave]; val = mask[s_] && mask[d_]; }
  else           { s_ = d_ = wave - EE; val = mask[s_] != 0; }
  if (!val) return;
  float e = S[s_] + D[d_];
  e = (e >= 0.f) ? e : 0.2f * e;
  float ex = __expf(e - m[d_]);
  if (lane == 0) atomicAdd(&den[d_], ex);
  const float* hs = hw + (size_t)s_ * HH;
  float* ad = acc + (size_t)d_ * HH;
#pragma unroll
  for (int j = 0; j < 8; j++) {
    int f = lane + 32 * j;
    atomicAdd(&ad[f], ex * hs[f]);
  }
}

// ---------------- normalize + bias + LayerNorm + ReLU, emit f32 and bf16 (wave per node) ----
__global__ __launch_bounds__(256) void k_norm(
    const float* __restrict__ acc, const float* __restrict__ den,
    const float* __restrict__ gb, const float* __restrict__ lg, const float* __restrict__ lb,
    float* __restrict__ hout, __bf16* __restrict__ hbf, int n) {
  int node = (blockIdx.x * 256 + threadIdx.x) >> 5;
  int lane = threadIdx.x & 31;
  if (node >= n) return;
  float dn = den[node];
  dn = (dn < 1e-16f) ? 1e-16f : dn;
  float v[8];
  float s = 0.f;
#pragma unroll
  for (int j = 0; j < 8; j++) {
    int f = lane + 32 * j;
    v[j] = acc[(size_t)node * HH + f] / dn + gb[f];
    s += v[j];
  }
  float mu = wave_sum(s) * (1.f / HH);
  float q = 0.f;
#pragma unroll
  for (int j = 0; j < 8; j++) { float t = v[j] - mu; q += t * t; }
  float var = wave_sum(q) * (1.f / HH);
  float rs = rsqrtf(var + 1e-5f);
#pragma unroll
  for (int j = 0; j < 8; j++) {
    int f = lane + 32 * j;
    float y = (v[j] - mu) * rs * lg[f] + lb[f];
    y = (y > 0.f) ? y : 0.f;
    hout[(size_t)node * HH + f] = y;
    hbf[(size_t)node * HH + f] = (__bf16)y;
  }
}

// ---------------- masked mean pool partials ----------------
__global__ __launch_bounds__(256) void k_pool(
    const float* __restrict__ h, const unsigned char* __restrict__ mask,
    float* __restrict__ pool, float* __restrict__ cnt, int n) {
  int f = threadIdx.x;
  int n0 = blockIdx.x * 256;
  float s = 0.f, c = 0.f;
  for (int i = 0; i < 256; i++) {
    int nd = n0 + i;
    if (nd >= n) break;
    float mk = mask[nd] ? 1.f : 0.f;
    s += mk * h[(size_t)nd * HH + f];
    c += mk;
  }
  atomicAdd(&pool[f], s);
  if (f == 0) atomicAdd(cnt, c);
}

// ---------------- head: MLP + LSTM step + projection (single block) ----------------
__global__ __launch_bounds__(256) void k_head(
    const float* __restrict__ pool, const float* __restrict__ cnt,
    const float* __restrict__ W1, const float* __restrict__ b1,
    const float* __restrict__ W2, const float* __restrict__ b2,
    const float* __restrict__ Wih, const float* __restrict__ Whh,
    const float* __restrict__ bih, const float* __restrict__ bhh,
    const float* __restrict__ pW, const float* __restrict__ pb,
    const float* __restrict__ h0, const float* __restrict__ c0,
    const float* __restrict__ cur, float* __restrict__ out) {
  __shared__ float sp[256], r1[256], rep[256], gates[2048], hn[512];
  int t = threadIdx.x;
  float c = cnt[0];
  c = (c < 1.f) ? 1.f : c;
  sp[t] = pool[t] / c;
  __syncthreads();

  float a = 0.f;
  for (int k = 0; k < 256; k++) a += sp[k] * W1[k * 256 + t];
  a += b1[t];
  r1[t] = (a > 0.f) ? a : 0.f;
  __syncthreads();

  float r = 0.f;
  for (int k = 0; k < 256; k++) r += r1[k] * W2[k * 256 + t];
  rep[t] = r + b2[t];
  __syncthreads();

#pragma unroll
  for (int q = 0; q < 8; q++) {
    int j = t + 256 * q;
    float g = bih[j] + bhh[j];
    for (int k = 0; k < 256; k++) g += rep[k] * Wih[j * 256 + k];
    for (int k = 0; k < 512; k++) g += h0[k] * Whh[j * 512 + k];
    gates[j] = g;
  }
  __syncthreads();

#pragma unroll
  for (int q = 0; q < 2; q++) {
    int u = t + 256 * q;
    float ig = gates[u], fg = gates[512 + u], gg = gates[1024 + u], og = gates[1536 + u];
    float si = 1.f / (1.f + __expf(-ig));
    float sf = 1.f / (1.f + __expf(-fg));
    float so = 1.f / (1.f + __expf(-og));
    float cn = sf * c0[u] + si * tanhf(gg);
    hn[u] = so * tanhf(cn);
  }
  __syncthreads();

#pragma unroll
  for (int q = 0; q < 4; q++) {
    int j = t + 256 * q;
    float o = pb[j] + cur[j];
    for (int k = 0; k < 512; k++) o += hn[k] * pW[k * 1024 + j];
    out[j] = o;
  }
}

// ---------------- workspace layout (bytes) ----------------
static constexpr size_t OFF_HBF  = 0;                      // 50000*256*2 = 25,600,000
static constexpr size_t OFF_HW   = 25600000;               // 50000*256*4 = 51,200,000 (also final h f32)
static constexpr size_t OFF_ACC  = 76800000;               // 51,200,000
static constexpr size_t OFF_WT   = 128000000;              // 256*256*2 = 131,072
static constexpr size_t OFF_S    = 128131072;              // 200,704 each
static constexpr size_t OFF_D    = OFF_S   + 200704;
static constexpr size_t OFF_M    = OFF_D   + 200704;
static constexpr size_t OFF_DEN  = OFF_M   + 200704;
static constexpr size_t OFF_POOL = OFF_DEN + 200704;       // 256 f32
static constexpr size_t OFF_CNT  = OFF_POOL + 1024;        // 1 f32

extern "C" void kernel_launch(void* const* d_in, const int* in_sizes, int n_in,
                              void* d_out, int out_size, void* d_ws, size_t ws_size,
                              hipStream_t stream) {
  const float* x = (const float*)d_in[0];
  const int* ei = (const int*)d_in[1];
  const unsigned char* mask = (const unsigned char*)d_in[3];
  const float* cur = (const float*)d_in[4];

  char* ws = (char*)d_ws;
  __bf16* hbf  = (__bf16*)(ws + OFF_HBF);
  float*  hw   = (float*)(ws + OFF_HW);
  float*  acc  = (float*)(ws + OFF_ACC);
  __bf16* wt   = (__bf16*)(ws + OFF_WT);
  float*  Sv   = (float*)(ws + OFF_S);
  float*  Dv   = (float*)(ws + OFF_D);
  float*  Mv   = (float*)(ws + OFF_M);
  float*  Den  = (float*)(ws + OFF_DEN);
  float*  Pool = (float*)(ws + OFF_POOL);
  float*  Cnt  = (float*)(ws + OFF_CNT);

  // x -> bf16 feature matrix
  k_cvt_bf16<<<(NN * HH + 255) / 256, 256, 0, stream>>>(x, hbf, NN * HH);

  for (int l = 0; l < 2; l++) {
    const float* W    = (const float*)d_in[5 + 6 * l];
    const float* asrc = (const float*)d_in[6 + 6 * l];
    const float* adst = (const float*)d_in[7 + 6 * l];
    const float* gb   = (const float*)d_in[8 + 6 * l];
    const float* lg   = (const float*)d_in[9 + 6 * l];
    const float* lb   = (const float*)d_in[10 + 6 * l];

    k_prep_wt<<<256, 256, 0, stream>>>(W, wt);
    {
      dim3 grid((NN + GM_ROWS - 1) / GM_ROWS, HH / GN_COLS);
      k_gemm_bf16<<<grid, 256, 0, stream>>>(hbf, wt, hw, NN);
    }
    k_sd<<<(NN + 7) / 8, 256, 0, stream>>>(hw, asrc, adst, Sv, Dv, NN);

    hipMemsetAsync(acc, 0, (size_t)NN * HH * sizeof(float), stream);
    hipMemsetAsync(Den, 0, (size_t)NN * sizeof(float), stream);
    k_init_m<<<(NN + 255) / 256, 256, 0, stream>>>(Mv, NN);

    k_edge_max<<<(ETOT + 255) / 256, 256, 0, stream>>>(ei, mask, Sv, Dv, Mv);
    k_fix_m<<<(NN + 255) / 256, 256, 0, stream>>>(Mv, NN);
    k_edge_acc<<<(ETOT + 7) / 8, 256, 0, stream>>>(ei, mask, Sv, Dv, Mv, hw, Den, acc);

    // acc/den + bias -> LN -> ReLU, write f32 (into hw) and bf16 (into hbf)
    k_norm<<<(NN + 7) / 8, 256, 0, stream>>>(acc, Den, gb, lg, lb, hw, hbf, NN);
  }

  hipMemsetAsync(Pool, 0, 1024 + 256, stream);  // pool + cnt
  k_pool<<<(NN + 255) / 256, 256, 0, stream>>>(hw, mask, Pool, Cnt, NN);

  k_head<<<1, 256, 0, stream>>>(
      Pool, Cnt,
      (const float*)d_in[17], (const float*)d_in[18],
      (const float*)d_in[19], (const float*)d_in[20],
      (const float*)d_in[21], (const float*)d_in[22],
      (const float*)d_in[23], (const float*)d_in[24],
      (const float*)d_in[25], (const float*)d_in[26],
      (const float*)d_in[27], (const float*)d_in[28],
      cur, (float*)d_out);
}
